// FullAttentionBlock_62612033241785
// MI455X (gfx1250) — matrix operand order repaired and checked by reading, not compiled, not run on hardware
//
#include <hip/hip_runtime.h>

// ---------------------------------------------------------------------------
// Problem constants (from reference): B=4, S=2048, hid=1024, H=8, KV=2, D=256
// ---------------------------------------------------------------------------
#define B_   4
#define S_   2048
#define HID_ 1024
#define H_   8
#define KV_  2
#define D_   256
#define G_   (H_ / KV_)          // 4 query heads per kv head
#define QGN  (2 * H_ * D_)       // 4096 cols of qg (q | gate)
#define KVN  (KV_ * D_)          // 512 cols of k / v

typedef __attribute__((ext_vector_type(16))) __bf16 v16bf;
typedef __attribute__((ext_vector_type(8)))  __bf16 bf16x8;
typedef __attribute__((ext_vector_type(8)))  float  v8f;

static __device__ inline __bf16 to_bf(float x)  { return (__bf16)x; }
static __device__ inline __bf16 to_bf(__bf16 x) { return x; }

// ---------------------------------------------------------------------------
// Fragment loaders per CDNA5 ISA 7.12.2 (16-bit, wave32).
// A 16x32 (MxK): lanes 0-15 row=lane,    K = {0..7, 16..23}
//                lanes 16-31 row=lane-16, K = {8..15, 24..31}
// B 32x16 (KxN): lanes 0-15 col=lane,    K = {0..7, 16..23}
//                lanes 16-31 col=lane-16, K = {8..15, 24..31}
// Tiles are stored with K contiguous (A: [m][k], B: [n][k]); all loads are
// aligned 16-byte ds_load_b128 / global b128.
// ---------------------------------------------------------------------------
static __device__ inline v16bf load_a_frag(const __bf16* base, int stride, int lane) {
  const __bf16* p = base + (size_t)(lane & 15) * stride + ((lane >> 4) << 3);
  bf16x8 a0 = *(const bf16x8*)(p);
  bf16x8 a1 = *(const bf16x8*)(p + 16);
  v16bf f;
#pragma unroll
  for (int i = 0; i < 8; ++i) { f[i] = a0[i]; f[i + 8] = a1[i]; }
  return f;
}

static __device__ inline v16bf load_b_frag(const __bf16* base, int stride, int lane) {
  const __bf16* p = base + (size_t)(lane & 15) * stride + ((lane >> 4) << 3);
  bf16x8 a0 = *(const bf16x8*)(p);
  bf16x8 a1 = *(const bf16x8*)(p + 16);
  v16bf f;
#pragma unroll
  for (int i = 0; i < 8; ++i) { f[i] = a0[i]; f[i + 8] = a1[i]; }
  return f;
}

// ---------------------------------------------------------------------------
// Async global->LDS staging (CDNA5 GLOBAL_LOAD_ASYNC_TO_LDS_B128, ASYNCcnt).
// Builtin signature (from hipcc diagnostic): (int4 AS1*, int4 AS3*, Ii, Ii).
// ---------------------------------------------------------------------------
#if __has_builtin(__builtin_amdgcn_global_load_async_to_lds_b128)
#define USE_ASYNC_K 1
typedef int v4i_t __attribute__((ext_vector_type(4)));
typedef __attribute__((address_space(1))) v4i_t g_v4i;
typedef __attribute__((address_space(3))) v4i_t l_v4i;

static __device__ inline void async_copy_b128(const __bf16* gsrc, __bf16* ldst) {
  __builtin_amdgcn_global_load_async_to_lds_b128(
      (g_v4i*)gsrc, (l_v4i*)ldst, 0, 0);
}
static __device__ inline void async_wait0() {
#if __has_builtin(__builtin_amdgcn_s_wait_asynccnt)
  __builtin_amdgcn_s_wait_asynccnt(0);
#else
  asm volatile("s_wait_asynccnt 0x0" ::: "memory");
#endif
}
#endif

// ---------------------------------------------------------------------------
// Generic WMMA GEMM: C[M,N] = A[M,K] * B[K,N]
//   A: row-major (fp32 or bf16), B: row-major fp32 (weights), C: fp32 or bf16
//   256 threads = 8 waves; block tile 128x128; K step 32.
//   wave w computes rows [(w/2)*32, +32) x cols [(w%2)*64, +64)  (2x4 WMMA tiles)
// ---------------------------------------------------------------------------
template <typename AT, typename OT>
__global__ __launch_bounds__(256)
void gemm_wmma_kernel(const AT* __restrict__ A, const float* __restrict__ B,
                      OT* __restrict__ C, int M, int N, int K) {
  __shared__ __attribute__((aligned(16))) __bf16 lds_a[128 * 32];   // [m][k]
  __shared__ __attribute__((aligned(16))) __bf16 lds_bt[128 * 32];  // [n][k]

  const int t    = threadIdx.x;
  const int lane = t & 31;
  const int wave = t >> 5;
  const int m0 = blockIdx.y * 128;
  const int n0 = blockIdx.x * 128;
  const int mb = (wave >> 1) * 32;
  const int nb = (wave & 1) * 64;

  v8f acc[2][4];
#pragma unroll
  for (int i = 0; i < 2; ++i)
#pragma unroll
    for (int j = 0; j < 4; ++j)
#pragma unroll
      for (int v = 0; v < 8; ++v) acc[i][j][v] = 0.0f;

  for (int k0 = 0; k0 < K; k0 += 32) {
    // Stage A tile 128x32 (convert to bf16)
#pragma unroll
    for (int i = 0; i < 16; ++i) {
      int idx = i * 256 + t;                 // 0..4095
      int r = idx >> 5, c = idx & 31;
      lds_a[idx] = to_bf(A[(size_t)(m0 + r) * K + (k0 + c)]);
    }
    // Stage B tile transposed: lds_bt[n][k], coalesced reads over n
#pragma unroll
    for (int i = 0; i < 16; ++i) {
      int idx = i * 256 + t;
      int kk = idx >> 7, n = idx & 127;
      lds_bt[n * 32 + kk] = to_bf(B[(size_t)(k0 + kk) * N + (n0 + n)]);
    }
    if (k0 + 32 < K)  // hint next weight panel into cache (global_prefetch_b8)
      __builtin_prefetch(&B[(size_t)(k0 + 32) * N + n0 + t], 0, 1);
    __syncthreads();

    v16bf af[2], bfv[4];
#pragma unroll
    for (int i = 0; i < 2; ++i)
      af[i] = load_a_frag(lds_a + (mb + 16 * i) * 32, 32, lane);
#pragma unroll
    for (int j = 0; j < 4; ++j)
      bfv[j] = load_b_frag(lds_bt + (nb + 16 * j) * 32, 32, lane);
#pragma unroll
    for (int i = 0; i < 2; ++i)
#pragma unroll
      for (int j = 0; j < 4; ++j)
        acc[i][j] = __builtin_amdgcn_wmma_f32_16x16x32_bf16(
            false, af[i], false, bfv[j], (short)0, acc[i][j], false, false);
    __syncthreads();
  }

  // Epilogue: C layout -> lane 0-15: col=lane, row=v; lane 16-31: col=lane-16, row=v+8
  const int rofs = (lane >> 4) << 3;
  const int cofs = lane & 15;
#pragma unroll
  for (int i = 0; i < 2; ++i)
#pragma unroll
    for (int j = 0; j < 4; ++j)
#pragma unroll
      for (int v = 0; v < 8; ++v) {
        int row = m0 + mb + 16 * i + v + rofs;
        int col = n0 + nb + 16 * j + cofs;
        C[(size_t)row * N + col] = (OT)acc[i][j][v];
      }
}

// ---------------------------------------------------------------------------
// Partial RoPE (first 64 dims of each head), in place on bf16 q (inside qg)
// and k. One thread per (row, head, pair i<32): pairs (i, i+32), inv_freq[i] =
// theta^(-2i/64), matching jnp.arange(0, 64, 2)/64.
// ---------------------------------------------------------------------------
__global__ void rope_kernel(__bf16* __restrict__ qg, __bf16* __restrict__ kb) {
  int tid = blockIdx.x * blockDim.x + threadIdx.x;
  const int total = B_ * S_ * (H_ + KV_) * 32;
  if (tid >= total) return;
  int i   = tid & 31;
  int h   = (tid >> 5) % (H_ + KV_);
  int row = tid / (32 * (H_ + KV_));
  int s   = row % S_;

  float invf = __powf(10000.0f, -(float)(2 * i) / 64.0f);
  float fr   = (float)s * invf;
  float c = __cosf(fr), sn = __sinf(fr);

  __bf16* base = (h < H_) ? (qg + (size_t)row * QGN + h * D_)
                          : (kb + (size_t)row * KVN + (h - H_) * D_);
  float x0 = (float)base[i];
  float x1 = (float)base[i + 32];
  base[i]      = (__bf16)(x0 * c - x1 * sn);
  base[i + 32] = (__bf16)(x1 * c + x0 * sn);
}

// ---------------------------------------------------------------------------
// Flash attention with WMMA + fused SiLU(gate) epilogue.
// Block = (b, kv) x 32 query positions; 8 waves = 4 g-heads x 2 row-tiles(16).
// K tile (32 keys x 256) staged with async global->LDS (ASYNCcnt) when the
// toolchain exposes the builtin; V^T tile transposed manually. Shared by all
// 4 heads of the kv group (GQA reuse).
// Per iteration per wave: 16 WMMAs (Q K^T) + 16 WMMAs (P V). 64 iterations.
// ---------------------------------------------------------------------------
__global__ __launch_bounds__(256)
void attn_kernel(const __bf16* __restrict__ qg, const __bf16* __restrict__ kb,
                 const __bf16* __restrict__ vb, __bf16* __restrict__ attn) {
  __shared__ __attribute__((aligned(16))) __bf16 lds_k[32 * 256];    // [key][d]
  __shared__ __attribute__((aligned(16))) __bf16 lds_vt[256 * 32];   // [d][key]
  __shared__ __attribute__((aligned(16))) __bf16 lds_p[8][16 * 32];  // per-wave P

  const int t = threadIdx.x, lane = t & 31, wave = t >> 5;
  const int qblk = blockIdx.x;                 // 64 blocks of 32 queries
  const int b = blockIdx.y / KV_, kv = blockIdx.y % KV_;
  const int g = wave >> 1, qsub = wave & 1;
  const int h = kv * G_ + g;                   // q head for (B,KV,G) grouping
  const size_t qrow0 = (size_t)b * S_ + qblk * 32 + qsub * 16;

  // Q fragments: 16 rows x 256 (8 k-steps of 32), straight from global bf16
  v16bf qf[8];
#pragma unroll
  for (int ks = 0; ks < 8; ++ks)
    qf[ks] = load_a_frag(qg + qrow0 * QGN + h * D_ + ks * 32, QGN, lane);

  v8f o[16];
#pragma unroll
  for (int tt = 0; tt < 16; ++tt)
#pragma unroll
    for (int v = 0; v < 8; ++v) o[tt][v] = 0.0f;
  float m[8], l[8];
#pragma unroll
  for (int v = 0; v < 8; ++v) { m[v] = -1e30f; l[v] = 0.0f; }

  const float scale = 0.0625f;  // 1/sqrt(256)

  for (int kt = 0; kt < S_ / 32; ++kt) {
    // Cooperative stage: 32x256 K (row-major) and V transposed -> [d][key]
#pragma unroll
    for (int j = 0; j < 4; ++j) {
      int c  = j * 256 + t;                    // 0..1023 chunks of 8 elems
      int n  = c >> 5;                         // key row within tile
      int co = (c & 31) * 8;                   // d offset
      size_t grow = (size_t)b * S_ + kt * 32 + n;
#if defined(USE_ASYNC_K)
      async_copy_b128(kb + grow * KVN + kv * D_ + co, lds_k + n * 256 + co);
#else
      bf16x8 kk = *(const bf16x8*)(kb + grow * KVN + kv * D_ + co);
      *(bf16x8*)(lds_k + n * 256 + co) = kk;
#endif
      bf16x8 vv = *(const bf16x8*)(vb + grow * KVN + kv * D_ + co);
#pragma unroll
      for (int e = 0; e < 8; ++e) lds_vt[(co + e) * 32 + n] = vv[e];
    }
#if defined(USE_ASYNC_K)
    async_wait0();
#endif
    __syncthreads();

    // Scores: 16 queries x 32 keys, K-dim = 256
    v8f sf[2];
#pragma unroll
    for (int j = 0; j < 2; ++j)
#pragma unroll
      for (int v = 0; v < 8; ++v) sf[j][v] = 0.0f;
#pragma unroll
    for (int ks = 0; ks < 8; ++ks) {
#pragma unroll
      for (int j = 0; j < 2; ++j) {
        v16bf bk = load_b_frag(lds_k + (j * 16) * 256 + ks * 32, 256, lane);
        sf[j] = __builtin_amdgcn_wmma_f32_16x16x32_bf16(
            false, qf[ks], false, bk, (short)0, sf[j], false, false);
      }
    }

    // Online softmax (rows live in 16-lane halves; xor 1/2/4/8 reduces a row)
    float alpha[8];
#pragma unroll
    for (int v = 0; v < 8; ++v) {
      float a0 = sf[0][v] * scale, a1 = sf[1][v] * scale;
      float tm = fmaxf(a0, a1);
      tm = fmaxf(tm, __shfl_xor(tm, 1));
      tm = fmaxf(tm, __shfl_xor(tm, 2));
      tm = fmaxf(tm, __shfl_xor(tm, 4));
      tm = fmaxf(tm, __shfl_xor(tm, 8));
      float mn = fmaxf(m[v], tm);
      alpha[v] = __expf(m[v] - mn);
      m[v] = mn;
      float p0 = __expf(a0 - mn), p1 = __expf(a1 - mn);
      sf[0][v] = p0; sf[1][v] = p1;
      float ts = p0 + p1;
      ts += __shfl_xor(ts, 1);
      ts += __shfl_xor(ts, 2);
      ts += __shfl_xor(ts, 4);
      ts += __shfl_xor(ts, 8);
      l[v] = l[v] * alpha[v] + ts;
    }
#pragma unroll
    for (int tt = 0; tt < 16; ++tt)
#pragma unroll
      for (int v = 0; v < 8; ++v) o[tt][v] *= alpha[v];

    // Re-fragment P via per-wave LDS (C-layout -> A-layout)
    {
      const int rofs = (lane >> 4) << 3, cofs = lane & 15;
#pragma unroll
      for (int j = 0; j < 2; ++j)
#pragma unroll
        for (int v = 0; v < 8; ++v)
          lds_p[wave][(v + rofs) * 32 + j * 16 + cofs] = (__bf16)sf[j][v];
    }
    __syncthreads();

    // O += P (16x32) * V (32x256)
    v16bf pf = load_a_frag(lds_p[wave], 32, lane);
#pragma unroll
    for (int tt = 0; tt < 16; ++tt) {
      v16bf vf = load_b_frag(lds_vt + (tt * 16) * 32, 32, lane);
      o[tt] = __builtin_amdgcn_wmma_f32_16x16x32_bf16(
          false, pf, false, vf, (short)0, o[tt], false, false);
    }
    __syncthreads();
  }

  // Epilogue: normalize, fuse SiLU(gate), write (B,S,H*D) layout as bf16
  const int rofs = (lane >> 4) << 3, cofs = lane & 15;
#pragma unroll
  for (int tt = 0; tt < 16; ++tt)
#pragma unroll
    for (int v = 0; v < 8; ++v) {
      int row = v + rofs;
      size_t grow = qrow0 + row;
      int col = tt * 16 + cofs;
      float gate = (float)qg[grow * QGN + H_ * D_ + h * D_ + col];
      float sig  = 1.0f / (1.0f + __expf(-gate));
      float val  = (o[tt][v] / l[v]) * gate * sig;
      attn[grow * (H_ * D_) + h * D_ + col] = (__bf16)val;
    }
}

// ---------------------------------------------------------------------------
// Host launcher
// ---------------------------------------------------------------------------
extern "C" void kernel_launch(void* const* d_in, const int* in_sizes, int n_in,
                              void* d_out, int out_size, void* d_ws, size_t ws_size,
                              hipStream_t stream) {
  (void)in_sizes; (void)n_in; (void)out_size; (void)ws_size;
  const float* hs = (const float*)d_in[0];   // (B,S,1024)
  const float* Wq = (const float*)d_in[1];   // (1024, 4096)
  const float* Wk = (const float*)d_in[2];   // (1024, 512)
  const float* Wv = (const float*)d_in[3];   // (1024, 512)
  const float* Wo = (const float*)d_in[4];   // (2048, 1024)
  float* out = (float*)d_out;                // (B,S,1024) fp32

  const int M = B_ * S_;                     // 8192 rows
  char* ws = (char*)d_ws;
  size_t off = 0;
  __bf16* qg = (__bf16*)(ws + off); off += (size_t)M * QGN * 2;          // 64 MB
  __bf16* kb = (__bf16*)(ws + off); off += (size_t)M * KVN * 2;          //  8 MB
  __bf16* vb = (__bf16*)(ws + off); off += (size_t)M * KVN * 2;          //  8 MB
  __bf16* at = (__bf16*)(ws + off);                                      // 32 MB

  // Projections (fp32 in -> bf16 out)
  gemm_wmma_kernel<float, __bf16>
      <<<dim3(QGN / 128, M / 128), 256, 0, stream>>>(hs, Wq, qg, M, QGN, HID_);
  gemm_wmma_kernel<float, __bf16>
      <<<dim3(KVN / 128, M / 128), 256, 0, stream>>>(hs, Wk, kb, M, KVN, HID_);
  gemm_wmma_kernel<float, __bf16>
      <<<dim3(KVN / 128, M / 128), 256, 0, stream>>>(hs, Wv, vb, M, KVN, HID_);

  // Partial RoPE on q (inside qg) and k, in place
  {
    int total = B_ * S_ * (H_ + KV_) * 32;
    rope_kernel<<<(total + 255) / 256, 256, 0, stream>>>(qg, kb);
  }

  // Flash attention + SiLU(gate) fusion -> bf16 (B,S,H*D)
  attn_kernel<<<dim3(S_ / 32, B_ * KV_), 256, 0, stream>>>(qg, kb, vb, at);

  // Output projection (bf16 in -> fp32 out)
  gemm_wmma_kernel<__bf16, float>
      <<<dim3(HID_ / 128, M / 128), 256, 0, stream>>>(at, Wo, out, M, HID_, 2048);
}
